// IndRNNModel_57939108823171
// MI455X (gfx1250) — matrix operand order repaired and checked by reading, not compiled, not run on hardware
//
#include <hip/hip_runtime.h>
#include <hip/hip_bf16.h>

#define B_ 256
#define T_ 256
#define D_ 512
#define H_ 512
#define O_ 256
#define NWG 32        // feature-slice workgroups
#define NF 16         // features per workgroup
#define NTHREADS 256  // 8 waves (wave32)
#define NWAVES 8
#define EPSV 1e-5f

typedef __bf16 bf16;
typedef __attribute__((ext_vector_type(8)))  __bf16 v8bf;
typedef __attribute__((ext_vector_type(16))) __bf16 v16bf;
typedef __attribute__((ext_vector_type(8)))  float  v8f;

static __device__ __forceinline__ v16bf cat16(v8bf lo, v8bf hi) {
  return __builtin_shufflevector(lo, hi, 0,1,2,3,4,5,6,7,8,9,10,11,12,13,14,15);
}

static __device__ __forceinline__ v8f wmma_bf16(v16bf a, v16bf b, v8f c) {
  return __builtin_amdgcn_wmma_f32_16x16x32_bf16(false, a, false, b, (short)0, c,
                                                 false, false);
}

// ---- grid-wide barrier over NWG persistent workgroups (L2 atomic counter) ----
static __device__ __forceinline__ void grid_barrier(unsigned* ctr, unsigned* phase) {
  __syncthreads();
  if (threadIdx.x == 0) {
    *phase += 1u;
    __hip_atomic_fetch_add(ctr, 1u, __ATOMIC_ACQ_REL, __HIP_MEMORY_SCOPE_AGENT);
    const unsigned target = (*phase) * NWG;
    while (__hip_atomic_load(ctr, __ATOMIC_ACQUIRE, __HIP_MEMORY_SCOPE_AGENT) < target) {
      __builtin_amdgcn_s_sleep(2);
    }
  }
  __syncthreads();
}

// ---- one IndRNN cell for this workgroup's 16-feature slice over all 256 rows ----
// out = BN2( relu( BN1(inp @ Wslice^T + b) + h*u ) ); h <- out; out -> global act
static __device__ __forceinline__ void run_cell(
    const bf16* __restrict__ aBase, int aStride,   // input rows [256 x 512] bf16
    const bf16* __restrict__ sW,                   // LDS weight slice [NF][512]
    const float* __restrict__ sPar,                // LDS params [6][16]: b,u,g1,be1,g2,be2
    float (&h)[2][8],                              // per-wave hidden regs (C/D layout)
    bf16* __restrict__ outBase, int colBase,       // output act [256 x 512] bf16
    float* sSum, float* sSq, float* sSc, float* sSh)
{
  const int tid  = threadIdx.x;
  const int lane = tid & 31;
  const int wave = tid >> 5;
  const int nl   = lane & 15;      // matrix column within tile (B/C/D), A row within tile
  const int hlf  = lane >> 4;      // lane half
  const int ka   = hlf * 8;        // A-fragment K offset within 32-chunk
  const int kb   = hlf * 16;       // unused; B fragment is loaded with ka, like A
  const int rof  = hlf * 8;        // C/D row offset

  v8f acc0 = {0.f,0.f,0.f,0.f,0.f,0.f,0.f,0.f};
  v8f acc1 = acc0;
  const int row0 = (2*wave + 0) * 16;
  const int row1 = (2*wave + 1) * 16;
  const bf16* aRow0 = aBase + (long)(row0 + nl) * aStride;
  const bf16* aRow1 = aBase + (long)(row1 + nl) * aStride;
  const bf16* wRow  = sW + nl * 512;

  for (int k = 0; k < 512; k += 32) {
    v16bf fb  = cat16(*(const v8bf*)(wRow  + k + ka), *(const v8bf*)(wRow  + k + ka + 16));
    v16bf fa0 = cat16(*(const v8bf*)(aRow0 + k + ka), *(const v8bf*)(aRow0 + k + ka + 16));
    v16bf fa1 = cat16(*(const v8bf*)(aRow1 + k + ka), *(const v8bf*)(aRow1 + k + ka + 16));
    acc0 = wmma_bf16(fa0, fb, acc0);
    acc1 = wmma_bf16(fa1, fb, acc1);
  }

  // bias + BN1 partial stats (each (m,n) element held exactly once in this wave)
  const float bb = sPar[0*16 + nl];
  float s = 0.f, q = 0.f;
  #pragma unroll
  for (int r = 0; r < 8; ++r) {
    acc0[r] += bb; acc1[r] += bb;
    s += acc0[r] + acc1[r];
    q += acc0[r]*acc0[r] + acc1[r]*acc1[r];
  }
  sSum[wave*32 + lane] = s;
  sSq [wave*32 + lane] = q;
  __syncthreads();
  if (tid < 16) {
    float S = 0.f, Q = 0.f;
    #pragma unroll
    for (int w = 0; w < NWAVES; ++w) {
      S += sSum[w*32 + tid] + sSum[w*32 + tid + 16];
      Q += sSq [w*32 + tid] + sSq [w*32 + tid + 16];
    }
    const float inv = 1.0f / (float)B_;
    const float mean = S * inv;
    const float var  = Q * inv - mean*mean;
    const float rs   = rsqrtf(var + EPSV);
    const float sc   = sPar[2*16 + tid] * rs;
    sSc[tid] = sc;
    sSh[tid] = sPar[3*16 + tid] - mean * sc;
  }
  __syncthreads();

  // apply BN1, recurrent u, relu; BN2 partial stats
  {
    const float sc = sSc[nl], sh = sSh[nl], uu = sPar[1*16 + nl];
    float s2 = 0.f, q2 = 0.f;
    #pragma unroll
    for (int r = 0; r < 8; ++r) {
      float v0 = fmaxf(acc0[r]*sc + sh + h[0][r]*uu, 0.f);
      float v1 = fmaxf(acc1[r]*sc + sh + h[1][r]*uu, 0.f);
      acc0[r] = v0; acc1[r] = v1;
      s2 += v0 + v1; q2 += v0*v0 + v1*v1;
    }
    sSum[wave*32 + lane] = s2;
    sSq [wave*32 + lane] = q2;
  }
  __syncthreads();
  if (tid < 16) {
    float S = 0.f, Q = 0.f;
    #pragma unroll
    for (int w = 0; w < NWAVES; ++w) {
      S += sSum[w*32 + tid] + sSum[w*32 + tid + 16];
      Q += sSq [w*32 + tid] + sSq [w*32 + tid + 16];
    }
    const float inv = 1.0f / (float)B_;
    const float mean = S * inv;
    const float var  = Q * inv - mean*mean;
    const float rs   = rsqrtf(var + EPSV);
    const float sc   = sPar[4*16 + tid] * rs;
    sSc[tid] = sc;
    sSh[tid] = sPar[5*16 + tid] - mean * sc;
  }
  __syncthreads();

  // apply BN2, update hidden regs, emit bf16 activation slice to global
  {
    const float sc = sSc[nl], sh = sSh[nl];
    #pragma unroll
    for (int r = 0; r < 8; ++r) {
      float v0 = acc0[r]*sc + sh;
      float v1 = acc1[r]*sc + sh;
      h[0][r] = v0; h[1][r] = v1;
      outBase[(row0 + r + rof)*H_ + colBase + nl] = (bf16)v0;
      outBase[(row1 + r + rof)*H_ + colBase + nl] = (bf16)v1;
    }
  }
}

__global__ __launch_bounds__(NTHREADS, 1)
void indrnn_persistent(
    const bf16* __restrict__ xbf,   const bf16* __restrict__ w0bf,
    const bf16* __restrict__ wmbf,  const bf16* __restrict__ wfcbf,
    bf16* __restrict__ actA, bf16* __restrict__ actB,
    unsigned* __restrict__ ctr,
    const float* __restrict__ b0,   const float* __restrict__ u0,
    const float* __restrict__ g1_0, const float* __restrict__ be1_0,
    const float* __restrict__ g2_0, const float* __restrict__ be2_0,
    const float* __restrict__ bm,   const float* __restrict__ um,
    const float* __restrict__ g1_m, const float* __restrict__ be1_m,
    const float* __restrict__ g2_m, const float* __restrict__ be2_m,
    const float* __restrict__ bfc,  float* __restrict__ out)
{
  __shared__ bf16  sW0[NF * 512];        // this wg's 16-row slice of W0
  __shared__ bf16  sWm[NF * 512];        // this wg's 16-row slice of Wm
  __shared__ float sPar0[6 * 16];
  __shared__ float sParM[6 * 16];
  __shared__ float sSum[NWAVES * 32];
  __shared__ float sSq [NWAVES * 32];
  __shared__ float sSc[16], sSh[16];
  __shared__ float sLog[16 * 272];       // head logits tile (pad vs bank conflicts)
  __shared__ float sLse[16];

  const int tid = threadIdx.x;
  const int wg  = blockIdx.x;
  const int colBase = wg * NF;

  // stage weight slices and parameter slices into LDS once
  for (int idx = tid; idx < NF * 512 / 8; idx += NTHREADS) {
    const int r  = idx >> 6;             // 64 v8bf chunks per 512-wide row
    const int c8 = idx & 63;
    ((v8bf*)sW0)[idx] = *(const v8bf*)(w0bf + (colBase + r)*512 + c8*8);
    ((v8bf*)sWm)[idx] = *(const v8bf*)(wmbf + (colBase + r)*512 + c8*8);
  }
  if (tid < 16) {
    const int c = colBase + tid;
    sPar0[0*16+tid]=b0[c];  sPar0[1*16+tid]=u0[c];
    sPar0[2*16+tid]=g1_0[c]; sPar0[3*16+tid]=be1_0[c];
    sPar0[4*16+tid]=g2_0[c]; sPar0[5*16+tid]=be2_0[c];
    sParM[0*16+tid]=bm[c];  sParM[1*16+tid]=um[c];
    sParM[2*16+tid]=g1_m[c]; sParM[3*16+tid]=be1_m[c];
    sParM[4*16+tid]=g2_m[c]; sParM[5*16+tid]=be2_m[c];
  }
  __syncthreads();

  float h0[2][8], h1[2][8], h2[2][8];
  #pragma unroll
  for (int t = 0; t < 2; ++t)
    #pragma unroll
    for (int r = 0; r < 8; ++r) { h0[t][r]=0.f; h1[t][r]=0.f; h2[t][r]=0.f; }

  unsigned phase = 0;
  for (int t = 0; t < T_; ++t) {
    // layer 0: x_t -> actA
    run_cell(xbf + (long)t * D_, T_ * D_, sW0, sPar0, h0, actA, colBase,
             sSum, sSq, sSc, sSh);
    grid_barrier(ctr, &phase);
    // middle layer 1 (shared params, own state): actA -> actB
    run_cell(actA, H_, sWm, sParM, h1, actB, colBase, sSum, sSq, sSc, sSh);
    grid_barrier(ctr, &phase);
    // middle layer 2: actB -> actA   (actA holds hm[-1] after final step)
    run_cell(actB, H_, sWm, sParM, h2, actA, colBase, sSum, sSq, sSc, sSh);
    grid_barrier(ctr, &phase);
  }

  // ---- head: logits = hm[-1] @ Wfc^T + bfc; log_softmax rows ----
  if (wg >= 16) return;
  const int rowBase = wg * 16;
  const int lane = tid & 31;
  const int wave = tid >> 5;
  const int nl   = lane & 15;
  const int hlf  = lane >> 4;
  const int ka   = hlf * 8;
  const int kb   = hlf * 16;
  const int rof  = hlf * 8;

  v8f l0 = {0.f,0.f,0.f,0.f,0.f,0.f,0.f,0.f};
  v8f l1 = l0;
  const int nt0 = 2*wave, nt1 = 2*wave + 1;
  const bf16* aRow = actA + (rowBase + nl) * H_;
  const bf16* wr0  = wfcbf + (nt0*16 + nl) * H_;
  const bf16* wr1  = wfcbf + (nt1*16 + nl) * H_;
  for (int k = 0; k < 512; k += 32) {
    v16bf fa  = cat16(*(const v8bf*)(aRow + k + ka), *(const v8bf*)(aRow + k + ka + 16));
    v16bf fb0 = cat16(*(const v8bf*)(wr0  + k + ka), *(const v8bf*)(wr0  + k + ka + 16));
    v16bf fb1 = cat16(*(const v8bf*)(wr1  + k + ka), *(const v8bf*)(wr1  + k + ka + 16));
    l0 = wmma_bf16(fa, fb0, l0);
    l1 = wmma_bf16(fa, fb1, l1);
  }
  const float bfv0 = bfc[nt0*16 + nl];
  const float bfv1 = bfc[nt1*16 + nl];
  #pragma unroll
  for (int r = 0; r < 8; ++r) {
    sLog[(r + rof)*272 + nt0*16 + nl] = l0[r] + bfv0;
    sLog[(r + rof)*272 + nt1*16 + nl] = l1[r] + bfv1;
  }
  __syncthreads();
  if (tid < 16) {
    const float* row = sLog + tid*272;
    float mx = -3.402823466e38f;
    for (int c = 0; c < O_; ++c) mx = fmaxf(mx, row[c]);
    float sum = 0.f;
    for (int c = 0; c < O_; ++c) sum += __expf(row[c] - mx);
    sLse[tid] = mx + __logf(sum);
  }
  __syncthreads();
  for (int idx = tid; idx < 16 * O_; idx += NTHREADS) {
    const int row = idx >> 8, col = idx & 255;
    out[(rowBase + row)*O_ + col] = sLog[row*272 + col] - sLse[row];
  }
}

// ---- prep kernels ----
__global__ void cvt_bf16_kernel(const float* __restrict__ src,
                                bf16* __restrict__ dst, int n) {
  int i = blockIdx.x * blockDim.x + threadIdx.x;
  const int stride = gridDim.x * blockDim.x;
  for (; i < n; i += stride) dst[i] = (bf16)src[i];
}
__global__ void init_ctr_kernel(unsigned* ctr) {
  if (threadIdx.x == 0 && blockIdx.x == 0) *ctr = 0u;
}

extern "C" void kernel_launch(void* const* d_in, const int* in_sizes, int n_in,
                              void* d_out, int out_size, void* d_ws, size_t ws_size,
                              hipStream_t stream) {
  const float* x     = (const float*)d_in[0];
  const float* W0    = (const float*)d_in[1];
  const float* b0    = (const float*)d_in[2];
  const float* u0    = (const float*)d_in[3];
  const float* g1_0  = (const float*)d_in[4];
  const float* be1_0 = (const float*)d_in[5];
  const float* g2_0  = (const float*)d_in[6];
  const float* be2_0 = (const float*)d_in[7];
  const float* Wm    = (const float*)d_in[8];
  const float* bm    = (const float*)d_in[9];
  const float* um    = (const float*)d_in[10];
  const float* g1_m  = (const float*)d_in[11];
  const float* be1_m = (const float*)d_in[12];
  const float* g2_m  = (const float*)d_in[13];
  const float* be2_m = (const float*)d_in[14];
  const float* Wfc   = (const float*)d_in[15];
  const float* bfc   = (const float*)d_in[16];
  float* out = (float*)d_out;

  char* ws = (char*)d_ws;
  bf16* xbf   = (bf16*)ws; ws += (size_t)B_*T_*D_*sizeof(bf16);
  bf16* w0bf  = (bf16*)ws; ws += (size_t)H_*D_*sizeof(bf16);
  bf16* wmbf  = (bf16*)ws; ws += (size_t)H_*H_*sizeof(bf16);
  bf16* wfcbf = (bf16*)ws; ws += (size_t)O_*H_*sizeof(bf16);
  bf16* actA  = (bf16*)ws; ws += (size_t)B_*H_*sizeof(bf16);
  bf16* actB  = (bf16*)ws; ws += (size_t)B_*H_*sizeof(bf16);
  unsigned* ctr = (unsigned*)ws;

  cvt_bf16_kernel<<<4096, 256, 0, stream>>>(x,   xbf,   B_*T_*D_);
  cvt_bf16_kernel<<<256,  256, 0, stream>>>(W0,  w0bf,  H_*D_);
  cvt_bf16_kernel<<<256,  256, 0, stream>>>(Wm,  wmbf,  H_*H_);
  cvt_bf16_kernel<<<128,  256, 0, stream>>>(Wfc, wfcbf, O_*H_);
  init_ctr_kernel<<<1, 64, 0, stream>>>(ctr);

  indrnn_persistent<<<NWG, NTHREADS, 0, stream>>>(
      xbf, w0bf, wmbf, wfcbf, actA, actB, ctr,
      b0, u0, g1_0, be1_0, g2_0, be2_0,
      bm, um, g1_m, be1_m, g2_m, be2_m,
      bfc, out);
}
